// EfficientDsConv_26207890440515
// MI455X (gfx1250) — matrix operand order repaired and checked by reading, not compiled, not run on hardware
//
#include <hip/hip_runtime.h>

// Problem constants (b, L, d, h) = (2, 8192, 1024, 16)
#define Bsz 2
#define Lseq 8192
#define Dm 1024
#define Hh 16
#define HD 64           // channels per head
#define BM 128          // output rows (t) per block
#define BK 32           // reduction (s) per k-step == WMMA K for bf16
#define KPAD_FRONT 128  // zeros in front of K => causal masking for free
#define KSTRIDE 8448    // 128 + 8192 + 128
#define RS 40           // uT LDS row stride in halfs (80B) -> conflict-free b128 reads

typedef __attribute__((ext_vector_type(16))) __bf16 v16bf;
typedef __attribute__((ext_vector_type(8)))  float  v8f;

__device__ __forceinline__ unsigned short f2bf(float f) {
  unsigned int x = __float_as_uint(f);
  x += 0x7fffu + ((x >> 16) & 1u);   // round-to-nearest-even
  return (unsigned short)(x >> 16);
}

// ---------------------------------------------------------------------------
// Kernel 1: K (L x h, f32) -> per-head bf16 filter with 128 leading zeros.
// ---------------------------------------------------------------------------
__global__ __launch_bounds__(256) void kprep_kernel(const float* __restrict__ w,
                                                    unsigned short* __restrict__ kpad) {
  const int idx = blockIdx.x * 256 + threadIdx.x;
  if (idx >= Hh * KSTRIDE) return;
  const int head = idx / KSTRIDE;
  const int pos  = idx - head * KSTRIDE;
  const int j    = pos - KPAD_FRONT;
  float v = 0.0f;
  if (j >= 0 && j < Lseq) v = w[(size_t)j * Hh + head];
  kpad[idx] = f2bf(v);
}

// ---------------------------------------------------------------------------
// Kernel 2: LayerNorm row (b*L rows of d=1024).
//   - residual = u * D   -> written to d_out (base for the conv accumulation)
//   - u (bf16)           -> workspace, feeds the WMMA GEMM
// ---------------------------------------------------------------------------
__global__ __launch_bounds__(256) void ln_kernel(const float* __restrict__ x,
                                                 const float* __restrict__ gamma,
                                                 const float* __restrict__ beta,
                                                 const float* __restrict__ paramD,
                                                 float* __restrict__ resid,
                                                 unsigned short* __restrict__ ubf) {
  const int row = blockIdx.x;                 // 0 .. b*L-1
  const size_t base = (size_t)row * Dm;
  const int tid = threadIdx.x;

  float4 v = *(const float4*)(x + base + tid * 4);
  float s  = v.x + v.y + v.z + v.w;
  float s2 = v.x * v.x + v.y * v.y + v.z * v.z + v.w * v.w;
  #pragma unroll
  for (int off = 16; off > 0; off >>= 1) {    // wave32 reduction
    s  += __shfl_down(s, off, 32);
    s2 += __shfl_down(s2, off, 32);
  }
  __shared__ float red[2][8];
  const int wid = tid >> 5;
  if ((tid & 31) == 0) { red[0][wid] = s; red[1][wid] = s2; }
  __syncthreads();
  if (tid == 0) {
    float a = 0.f, b = 0.f;
    #pragma unroll
    for (int w = 0; w < 8; ++w) { a += red[0][w]; b += red[1][w]; }
    red[0][0] = a * (1.0f / Dm);
    red[1][0] = b * (1.0f / Dm);
  }
  __syncthreads();
  const float mu   = red[0][0];
  const float var  = red[1][0] - mu * mu;
  const float rstd = rsqrtf(var + 1e-5f);

  const int c = tid * 4;
  float4 g  = *(const float4*)(gamma + c);
  float4 bb = *(const float4*)(beta + c);
  float4 dd = *(const float4*)(paramD + c);
  float u0 = (v.x - mu) * rstd * g.x + bb.x;
  float u1 = (v.y - mu) * rstd * g.y + bb.y;
  float u2 = (v.z - mu) * rstd * g.z + bb.z;
  float u3 = (v.w - mu) * rstd * g.w + bb.w;

  float4 r = { u0 * dd.x, u1 * dd.y, u2 * dd.z, u3 * dd.w };
  *(float4*)(resid + base + c) = r;

  uint2 p;
  p.x = (unsigned)f2bf(u0) | ((unsigned)f2bf(u1) << 16);
  p.y = (unsigned)f2bf(u2) | ((unsigned)f2bf(u3) << 16);
  *(uint2*)(ubf + base + c) = p;
}

// ---------------------------------------------------------------------------
// Kernel 3: causal Toeplitz GEMM via v_wmma_f32_16x16x32_bf16, double-buffered.
// Block = 256 threads (8 waves). Block tile: 128 t-rows x 64 channels (1 head).
// Wave m owns t-rows [t0+16m, t0+16m+16) x all 64 channels -> 4 accumulators.
// Reduction mapping (reversed): s = s0 + 31 - k', so
//   A[i,k'] = Kh[t - s] = KS[16m + i + k']   (2 unaligned ds_load_b128 / wave)
//   B[k',n] = u[s0+31-k', c0+n] = uT[n][k']  (aligned ds_load_b128 pairs)
// Causality: Kpad's 128 leading zeros make K[t-s]=0 whenever s>t.
// Pipeline: while the WMMAs chew on buffer `cur`, the next tile's global loads
// are already in flight; they are committed to buffer `cur^1` after the WMMAs,
// and one barrier per iteration publishes them.
// ---------------------------------------------------------------------------
__global__ __launch_bounds__(256) void conv_kernel(const unsigned short* __restrict__ ubf,
                                                   const unsigned short* __restrict__ kpad,
                                                   float* __restrict__ out) {
  __shared__ unsigned short sKS[2][192];                                  // K sliding windows
  __shared__ __attribute__((aligned(16))) unsigned short sUT[2][HD * RS]; // u tiles (T + s-reversed)

  const int t0    = blockIdx.x * BM;
  const int batch = blockIdx.y >> 4;
  const int head  = blockIdx.y & 15;
  const int c0    = head * HD;
  const int tid   = threadIdx.x;
  const int wave  = tid >> 5;
  const int lane  = tid & 31;
  const int li    = lane & 15;   // row i (A) / column n (B,D)
  const int hf    = lane >> 4;   // lane-half select per WMMA layout

  const unsigned short* uB = ubf  + (size_t)batch * Lseq * Dm;
  const unsigned short* kh = kpad + (size_t)head * KSTRIDE;

  v8f acc[4] = {v8f{}, v8f{}, v8f{}, v8f{}};

  const int nk = (t0 + BM) / BK;       // only s-tiles with s0 <= t_max (causal)
  const int sl = tid >> 3;             // staging: s-row 0..31
  const int cg = (tid & 7) << 3;       // staging: channel group 0,8,..,56
  const int kk = 31 - sl;              // s-reversal absorbed at staging time
  const int rowoff = (wave << 4) + li; // A row within block (0..127)

  // ---- prologue: stage tile 0 into buffer 0 ----
  {
    const int kbase = (KPAD_FRONT - 31) + t0;      // s0 = 0
    if (tid < 160) sKS[0][tid] = kh[kbase + tid];
    const unsigned short* src = uB + (size_t)sl * Dm + c0 + cg;
    union { uint4 v; unsigned short h[8]; } ld;
    ld.v = *(const uint4*)src;
    #pragma unroll
    for (int q = 0; q < 8; ++q) sUT[0][(cg + q) * RS + kk] = ld.h[q];
  }

  for (int ks = 0; ks < nk; ++ks) {
    const int cur = ks & 1;
    const int nxt = cur ^ 1;
    const bool have_next = (ks + 1 < nk);

    __syncthreads();   // publish buffer `cur`; all reads of `nxt` (iter ks-1) done

    // ---- issue next tile's global loads NOW (latency hidden by WMMAs) ----
    union { uint4 v; unsigned short h[8]; } nu; nu.v = uint4{0, 0, 0, 0};
    unsigned short nks = 0;
    const unsigned short* src = uB + (size_t)((ks + 1) * BK + sl) * Dm + c0 + cg;
    if (have_next) {
      nu.v = *(const uint4*)src;                                       // global_load_b128
      if (tid < 160) nks = kh[(KPAD_FRONT - 31) + t0 - (ks + 1) * BK + tid];
      if (ks + 2 < nk) __builtin_prefetch(src + (size_t)BK * Dm, 0, 0); // tile ks+2
    }

    // ---- A operand: 16 contiguous+contiguous halfs -> 2 ds_load_b128 ----
    v16bf a;
    #pragma unroll
    for (int e = 0; e < 16; ++e) {
      const int kp = e + (hf << 4);
      union { unsigned short u; __bf16 b; } cv;
      cv.u = sKS[cur][rowoff + kp];
      a[e] = cv.b;
    }

    // ---- B operands: batch all 8 ds_load_b128, then burst 4 WMMAs ----
    v16bf btile[4];
    #pragma unroll
    for (int j = 0; j < 4; ++j) {
      union { struct { uint4 lo, hi; } q; v16bf v; } bv;
      const unsigned short* bp = &sUT[cur][(((j << 4) + li) * RS) + (hf << 4)];
      bv.q.lo = *(const uint4*)bp;
      bv.q.hi = *(const uint4*)(bp + 8);
      btile[j] = bv.v;
    }
    #pragma unroll
    for (int j = 0; j < 4; ++j) {
      acc[j] = __builtin_amdgcn_wmma_f32_16x16x32_bf16(
          /*neg_a=*/false, a, /*neg_b=*/false, btile[j],
          /*c_mod=*/(short)0, acc[j], /*reuse_a=*/false, /*reuse_b=*/false);
    }

    // ---- commit next tile into the other buffer ----
    if (have_next) {
      if (tid < 160) sKS[nxt][tid] = nks;
      #pragma unroll
      for (int q = 0; q < 8; ++q) sUT[nxt][(cg + q) * RS + kk] = nu.h[q];
    }
  }

  // D layout: VGPR q -> M = q + 8*hf, N = li. Add residual already in d_out.
  const size_t ob = (size_t)batch * Lseq * Dm;
  #pragma unroll
  for (int j = 0; j < 4; ++j) {
    const int c = c0 + (j << 4) + li;
    #pragma unroll
    for (int q = 0; q < 8; ++q) {
      const int t = t0 + (wave << 4) + q + (hf << 3);
      const size_t o = ob + (size_t)t * Dm + c;
      out[o] = out[o] + acc[j][q];
    }
  }
}

// ---------------------------------------------------------------------------
extern "C" void kernel_launch(void* const* d_in, const int* in_sizes, int n_in,
                              void* d_out, int out_size, void* d_ws, size_t ws_size,
                              hipStream_t stream) {
  const float* x      = (const float*)d_in[0];
  const float* weight = (const float*)d_in[1];
  const float* paramD = (const float*)d_in[2];
  const float* gamma  = (const float*)d_in[3];
  const float* beta   = (const float*)d_in[4];
  float* out = (float*)d_out;

  unsigned short* ubf  = (unsigned short*)d_ws;                        // 32 MiB u (bf16)
  unsigned short* kpad = (unsigned short*)((char*)d_ws +
                          (size_t)Bsz * Lseq * Dm * sizeof(unsigned short)); // ~264 KiB

  kprep_kernel<<<(Hh * KSTRIDE + 255) / 256, 256, 0, stream>>>(weight, kpad);
  ln_kernel<<<Bsz * Lseq, 256, 0, stream>>>(x, gamma, beta, paramD, out, ubf);
  conv_kernel<<<dim3(Lseq / BM, Bsz * Hh), 256, 0, stream>>>(ubf, kpad, out);
}
